// Block_29085518528833
// MI455X (gfx1250) — matrix operand order repaired and checked by reading, not compiled, not run on hardware
//
#include <hip/hip_runtime.h>
#include <hip/hip_bf16.h>
#include <stdint.h>

// ---------- types ----------
typedef __attribute__((ext_vector_type(16))) __bf16 v16bf;
typedef __attribute__((ext_vector_type(8)))  float  v8f;
typedef unsigned short u16;
typedef __attribute__((ext_vector_type(8))) u16 u16x8;

union Frag16 {            // one WMMA 16-bf16 operand = 2x 16-byte chunks
    v16bf bf;
    u16x8 h[2];
};

__device__ __forceinline__ u16 f2bf(float f) {
    unsigned int u = __float_as_uint(f);
    u = (u + 0x7FFFu + ((u >> 16) & 1u)) >> 16;
    return (u16)u;
}
__device__ __forceinline__ float bf2f(u16 h) {
    return __uint_as_float(((unsigned int)h) << 16);
}
__device__ __forceinline__ float hswish(float x) {
    float t = fminf(fmaxf(x + 3.0f, 0.0f), 6.0f);
    return x * t * (1.0f / 6.0f);
}

#define WMMA_BF16(A, B, C) \
    __builtin_amdgcn_wmma_f32_16x16x32_bf16(false, (A), false, (B), (short)0, (C), false, false)

// CDNA5 async global->LDS copy (ASYNCcnt-tracked, no VGPR round trip)
__device__ __forceinline__ void async_ld_b128(unsigned lds_off, const u16* g) {
    asm volatile("global_load_async_to_lds_b128 %0, %1, off"
                 :: "v"(lds_off), "v"((unsigned long long)(size_t)g)
                 : "memory");
}
__device__ __forceinline__ void wait_async(bool one_batch_in_flight) {
    if (one_batch_in_flight)
        asm volatile("s_wait_asynccnt 0x4" ::: "memory");
    else
        asm volatile("s_wait_asynccnt 0x0" ::: "memory");
}

// ---------- constants ----------
#define Bsz    4
#define Hh     96
#define Ww     96
#define Nn     9216         // H*W
#define Cc     512
#define HEADS  8
#define HD     64
#define HID    2048
#define Lpool  141

// ---------- elementwise f32 -> bf16 ----------
__global__ __launch_bounds__(256) void cvt_bf16_kernel(const float* __restrict__ in,
                                                       u16* __restrict__ out, int n) {
    int i = blockIdx.x * 256 + threadIdx.x;
    if (i < n) out[i] = f2bf(in[i]);
}

// ---------- LayerNorm over C=512, f32 in -> bf16 out ----------
__global__ __launch_bounds__(256) void ln_kernel(const float* __restrict__ in,
                                                 const float* __restrict__ g,
                                                 const float* __restrict__ bb,
                                                 u16* __restrict__ out) {
    __shared__ float red[256];
    const int row = blockIdx.x;
    const int t = threadIdx.x;
    const float* x = in + (size_t)row * Cc;
    float a0 = x[t], a1 = x[t + 256];
    red[t] = a0 + a1;
    __syncthreads();
    for (int off = 128; off > 0; off >>= 1) {
        if (t < off) red[t] += red[t + off];
        __syncthreads();
    }
    float mu = red[0] * (1.0f / Cc);
    __syncthreads();
    float d0 = a0 - mu, d1 = a1 - mu;
    red[t] = d0 * d0 + d1 * d1;
    __syncthreads();
    for (int off = 128; off > 0; off >>= 1) {
        if (t < off) red[t] += red[t + off];
        __syncthreads();
    }
    float rs = rsqrtf(red[0] * (1.0f / Cc) + 1e-5f);
    out[(size_t)row * Cc + t]       = f2bf(d0 * rs * g[t] + bb[t]);
    out[(size_t)row * Cc + t + 256] = f2bf(d1 * rs * g[t + 256] + bb[t + 256]);
}

// ---------- generic bf16 WMMA GEMM:  out[m,n] = sum_k A[m,k] * W[n,k] (+epilogue) ----------
// 128x128 block tile, 8 waves x (32x64) sub-tiles, K-step 32,
// double-buffered LDS filled with CDNA5 async global->LDS loads.
// EPI: 0 = bf16 store; 1 = bf16 store of hardswish(acc+bias); 2 = f32 store of acc+bias+resid
template <int EPI>
__global__ __launch_bounds__(256) void gemm_bf16_kernel(const u16* __restrict__ A,
                                                        const u16* __restrict__ W,
                                                        const float* __restrict__ bias,
                                                        const float* __restrict__ resid,
                                                        void* __restrict__ outp,
                                                        int M, int N, int K) {
    __shared__ u16 As[2][128][40];   // padded stride 80B (16B-aligned rows)
    __shared__ u16 Ws[2][128][40];

    const int tid  = threadIdx.x;
    const int m0   = blockIdx.y * 128;
    const int n0   = blockIdx.x * 128;
    const int w    = tid >> 5;
    const int lane = tid & 31;
    const int lo   = lane & 15;
    const int hi   = lane >> 4;
    const int wm   = (w >> 1) * 32;    // wave row base: 0,32,64,96
    const int wn   = (w & 1) * 64;     // wave col base: 0,64

    // per-thread staging assignment: tile = 128 rows x 32 cols = 512 x (8-elem chunks)
    const int r0 = tid >> 1,           s0 = (tid & 1) * 8;          // chunks 0..255
    const int r1 = (tid + 256) >> 1,   s1 = ((tid + 256) & 1) * 8;  // chunks 256..511
    // clamp out-of-range A rows to a valid row: keeps async-instruction count uniform
    // (garbage only feeds output rows >= M, which the store guard drops)
    const int ga0 = (m0 + r0 < M) ? (m0 + r0) : (M - 1);
    const int ga1 = (m0 + r1 < M) ? (m0 + r1) : (M - 1);

    const int KT = K >> 5;   // number of 32-wide K tiles

    auto issue = [&](int buf, int kt) {
        const int k0 = kt << 5;
        async_ld_b128((unsigned)(size_t)&As[buf][r0][s0], A + (size_t)ga0 * K + k0 + s0);
        async_ld_b128((unsigned)(size_t)&As[buf][r1][s1], A + (size_t)ga1 * K + k0 + s1);
        async_ld_b128((unsigned)(size_t)&Ws[buf][r0][s0], W + (size_t)(n0 + r0) * K + k0 + s0);
        async_ld_b128((unsigned)(size_t)&Ws[buf][r1][s1], W + (size_t)(n0 + r1) * K + k0 + s1);
    };

    v8f acc[2][4];
#pragma unroll
    for (int ti = 0; ti < 2; ++ti)
#pragma unroll
        for (int tj = 0; tj < 4; ++tj) acc[ti][tj] = (v8f){0, 0, 0, 0, 0, 0, 0, 0};

    issue(0, 0);
    if (KT > 1) issue(1, 1);

    for (int kt = 0; kt < KT; ++kt) {
        const int buf = kt & 1;
        wait_async(kt < KT - 1);    // my batch for `buf` is complete (in-order)
        __syncthreads();            // everyone's tile data visible

        Frag16 a[2], b[4];
#pragma unroll
        for (int ti = 0; ti < 2; ++ti) {
            const u16* ar = &As[buf][wm + ti * 16 + lo][0];
            a[ti].h[0] = *(const u16x8*)(ar + hi * 8);
            a[ti].h[1] = *(const u16x8*)(ar + 16 + hi * 8);
        }
#pragma unroll
        for (int tj = 0; tj < 4; ++tj) {
            const u16* br = &Ws[buf][wn + tj * 16 + lo][0];
            b[tj].h[0] = *(const u16x8*)(br + hi * 8);
            b[tj].h[1] = *(const u16x8*)(br + 16 + hi * 8);
        }
#pragma unroll
        for (int ti = 0; ti < 2; ++ti)
#pragma unroll
            for (int tj = 0; tj < 4; ++tj)
                acc[ti][tj] = WMMA_BF16(a[ti].bf, b[tj].bf, acc[ti][tj]);

        __syncthreads();            // all waves done reading `buf`
        if (kt + 2 < KT) issue(buf, kt + 2);
    }

#pragma unroll
    for (int ti = 0; ti < 2; ++ti) {
#pragma unroll
        for (int tj = 0; tj < 4; ++tj) {
            int gn = n0 + wn + tj * 16 + lo;
#pragma unroll
            for (int v = 0; v < 8; ++v) {
                int gm = m0 + wm + ti * 16 + hi * 8 + v;
                if (gm >= M) continue;
                size_t idx = (size_t)gm * N + gn;
                float x = acc[ti][tj][v];
                if (EPI == 0) {
                    ((u16*)outp)[idx] = f2bf(x);
                } else if (EPI == 1) {
                    x += bias[gn];
                    ((u16*)outp)[idx] = f2bf(hswish(x));
                } else {
                    ((float*)outp)[idx] = x + bias[gn] + resid[idx];
                }
            }
        }
    }
}

// ---------- pooled-cell helpers ----------
__device__ __forceinline__ void cell_decode(int l, int& i, int& oh, int& iy, int& ix, int& base) {
    if (l < 64)       { i = 0; oh = 8; base = 0;   l -= 0; }
    else if (l < 100) { i = 1; oh = 6; base = 64;  l -= 64; }
    else if (l < 125) { i = 2; oh = 5; base = 100; l -= 100; }
    else              { i = 3; oh = 4; base = 125; l -= 125; }
    iy = l / oh;
    ix = l % oh;
}

// adaptive average pool (exact torch bin rule), xn bf16 [B,N,C] -> pools_raw f32 [B,L,C]
__global__ __launch_bounds__(256) void pool_sum_kernel(const u16* __restrict__ xn,
                                                       float* __restrict__ pools_raw) {
    const int l = blockIdx.x, b = blockIdx.y;
    int i, oh, iy, ix, base;
    cell_decode(l, i, oh, iy, ix, base);
    int sh = (iy * Hh) / oh, eh = ((iy + 1) * Hh + oh - 1) / oh;
    int sw = (ix * Ww) / oh, ew = ((ix + 1) * Ww + oh - 1) / oh;
    float inv_area = 1.0f / (float)((eh - sh) * (ew - sw));
    const int c = threadIdx.x;
    float a0 = 0.f, a1 = 0.f;
    for (int y = sh; y < eh; ++y)
        for (int x = sw; x < ew; ++x) {
            const u16* p = xn + ((size_t)b * Nn + y * Ww + x) * Cc;
            a0 += bf2f(p[c]);
            a1 += bf2f(p[c + 256]);
        }
    float* dst = pools_raw + ((size_t)b * Lpool + l) * Cc;
    dst[c]       = a0 * inv_area;
    dst[c + 256] = a1 * inv_area;
}

// pool = pool + dwconv3x3(pool) + bias  (depthwise per ratio grid, zero-padded)
__global__ __launch_bounds__(256) void pool_dw_kernel(const float* __restrict__ pr,
                                                      const float* __restrict__ dw,
                                                      const float* __restrict__ db,
                                                      float* __restrict__ out) {
    const int l = blockIdx.x, b = blockIdx.y;
    int i, oh, iy, ix, base;
    cell_decode(l, i, oh, iy, ix, base);
    const int t = threadIdx.x;
#pragma unroll
    for (int half = 0; half < 2; ++half) {
        int c = t + half * 256;
        float acc = db[i * Cc + c];
        const float* wv = dw + ((size_t)(i * Cc + c)) * 9;
#pragma unroll
        for (int ky = 0; ky < 3; ++ky) {
            int ny = iy + ky - 1;
            if (ny < 0 || ny >= oh) continue;
#pragma unroll
            for (int kx = 0; kx < 3; ++kx) {
                int nx = ix + kx - 1;
                if (nx < 0 || nx >= oh) continue;
                acc += wv[ky * 3 + kx] *
                       pr[((size_t)b * Lpool + base + ny * oh + nx) * Cc + c];
            }
        }
        size_t idx = ((size_t)b * Lpool + l) * Cc + c;
        out[idx] = pr[idx] + acc;
    }
}

// ---------- attention: per block 64 Q rows for one (b,h) ----------
__global__ __launch_bounds__(128) void attn_kernel(const u16* __restrict__ qb,
                                                   const u16* __restrict__ kvb,
                                                   u16* __restrict__ outb) {
    __shared__ u16   Ks[144][72];       // K rows, zero-padded past L
    __shared__ u16   Vts[64][168];      // V transposed [d][l], l padded to 160
    __shared__ float Ss[4][16][148];    // per-wave scores
    __shared__ u16   Ps[4][16][168];    // per-wave probabilities bf16

    const int b = blockIdx.z, h = blockIdx.y, n0 = blockIdx.x * 64;
    const int tid = threadIdx.x;
    const int w = tid >> 5, lane = tid & 31, lo = lane & 15, hi = lane >> 4;

    for (int i = tid; i < 144 * 8; i += 128) {
        int l = i >> 3, seg = (i & 7) * 8;
        u16x8 v = {0, 0, 0, 0, 0, 0, 0, 0};
        if (l < Lpool)
            v = *(const u16x8*)(kvb + ((size_t)(b * Lpool + l)) * 1024 + h * HD + seg);
        *(u16x8*)(&Ks[l][seg]) = v;
    }
    for (int i = tid; i < 160 * 64; i += 128) {
        int l = i >> 6, d = i & 63;
        u16 v = 0;
        if (l < Lpool)
            v = kvb[((size_t)(b * Lpool + l)) * 1024 + 512 + h * HD + d];
        Vts[d][l] = v;
    }
    __syncthreads();

    // Q A-fragments (two K-chunks of 32)
    Frag16 qa0, qa1;
    {
        const u16* qrow = qb + ((size_t)b * Nn + n0 + w * 16 + lo) * Cc + h * HD;
        qa0.h[0] = *(const u16x8*)(qrow + hi * 8);
        qa0.h[1] = *(const u16x8*)(qrow + 16 + hi * 8);
        qa1.h[0] = *(const u16x8*)(qrow + 32 + hi * 8);
        qa1.h[1] = *(const u16x8*)(qrow + 48 + hi * 8);
    }

    // S = Q * K^T over 9 column tiles
    for (int t = 0; t < 9; ++t) {
        v8f acc = {0, 0, 0, 0, 0, 0, 0, 0};
        const u16* kr = &Ks[t * 16 + lo][0];
        Frag16 bk;
        bk.h[0] = *(const u16x8*)(kr + hi * 8);
        bk.h[1] = *(const u16x8*)(kr + 16 + hi * 8);
        acc = WMMA_BF16(qa0.bf, bk.bf, acc);
        bk.h[0] = *(const u16x8*)(kr + 32 + hi * 8);
        bk.h[1] = *(const u16x8*)(kr + 48 + hi * 8);
        acc = WMMA_BF16(qa1.bf, bk.bf, acc);
#pragma unroll
        for (int v = 0; v < 8; ++v) Ss[w][hi * 8 + v][t * 16 + lo] = acc[v];
    }
    __syncthreads();

    // softmax over L=141 (one lane per row)
    if (lane < 16) {
        float* srow = &Ss[w][lane][0];
        u16* prow = &Ps[w][lane][0];
        float m = -1e30f;
        for (int l = 0; l < Lpool; ++l) {
            float s = srow[l] * 0.125f;   // hd^-0.5 = 1/8
            srow[l] = s;
            m = fmaxf(m, s);
        }
        float sum = 0.f;
        for (int l = 0; l < Lpool; ++l) {
            float e = __expf(srow[l] - m);
            srow[l] = e;
            sum += e;
        }
        float inv = 1.0f / sum;
        for (int l = 0; l < Lpool; ++l) prow[l] = f2bf(srow[l] * inv);
        for (int l = Lpool; l < 160; ++l) prow[l] = 0;
    }
    __syncthreads();

    // O = P * V  (K padded to 160 -> 5 chunks)
    for (int t = 0; t < 4; ++t) {
        v8f acc = {0, 0, 0, 0, 0, 0, 0, 0};
#pragma unroll
        for (int kc = 0; kc < 5; ++kc) {
            Frag16 pa, bv;
            const u16* pr = &Ps[w][lo][kc * 32];
            pa.h[0] = *(const u16x8*)(pr + hi * 8);
            pa.h[1] = *(const u16x8*)(pr + 16 + hi * 8);
            const u16* vr = &Vts[t * 16 + lo][kc * 32];
            bv.h[0] = *(const u16x8*)(vr + hi * 8);
            bv.h[1] = *(const u16x8*)(vr + 16 + hi * 8);
            acc = WMMA_BF16(pa.bf, bv.bf, acc);
        }
#pragma unroll
        for (int v = 0; v < 8; ++v) {
            size_t row = (size_t)b * Nn + n0 + w * 16 + hi * 8 + v;
            outb[row * Cc + h * HD + t * 16 + lo] = f2bf(acc[v]);
        }
    }
}

// ---------- depthwise 3x3 over [B,N,HID] bf16, + bias + hardswish ----------
__global__ __launch_bounds__(256) void dwconv_kernel(const u16* __restrict__ t1,
                                                     const float* __restrict__ cw,
                                                     const float* __restrict__ cb,
                                                     u16* __restrict__ t2) {
    __shared__ u16 tile[6][96][64];
    __shared__ float wsm[64][9];
    __shared__ float bsm[64];
    const int c0 = blockIdx.x * 64;
    const int y0 = blockIdx.y * 4;
    const int b  = blockIdx.z;
    const int tid = threadIdx.x;

    for (int i = tid; i < 64 * 9; i += 256) wsm[i / 9][i % 9] = cw[(size_t)(c0 + i / 9) * 9 + (i % 9)];
    for (int i = tid; i < 64; i += 256) bsm[i] = cb[c0 + i];

    for (int r = 0; r < 6; ++r) {
        int y = y0 - 1 + r;
        for (int i = tid; i < 768; i += 256) {
            int x = i >> 3, seg = (i & 7) * 8;
            u16x8 v = {0, 0, 0, 0, 0, 0, 0, 0};
            if (y >= 0 && y < Hh)
                v = *(const u16x8*)(t1 + ((size_t)b * Nn + y * Ww + x) * HID + c0 + seg);
            *(u16x8*)(&tile[r][x][seg]) = v;
        }
    }
    __syncthreads();

    for (int oi = tid; oi < 4 * 96 * 64; oi += 256) {
        int c  = oi & 63;
        int x  = (oi >> 6) % 96;
        int dy = oi / (96 * 64);
        float acc = bsm[c];
#pragma unroll
        for (int ky = 0; ky < 3; ++ky) {
#pragma unroll
            for (int kx = 0; kx < 3; ++kx) {
                int xx = x + kx - 1;
                if (xx >= 0 && xx < Ww)
                    acc += wsm[c][ky * 3 + kx] * bf2f(tile[dy + ky][xx][c]);
            }
        }
        t2[((size_t)b * Nn + (y0 + dy) * Ww + x) * HID + c0 + c] = f2bf(hswish(acc));
    }
}

// ---------- launcher ----------
extern "C" void kernel_launch(void* const* d_in, const int* in_sizes, int n_in,
                              void* d_out, int out_size, void* d_ws, size_t ws_size,
                              hipStream_t stream) {
    (void)in_sizes; (void)n_in; (void)out_size; (void)ws_size;
    const float* x        = (const float*)d_in[0];
    const float* norm1_g  = (const float*)d_in[1];
    const float* norm1_b  = (const float*)d_in[2];
    const float* q_w      = (const float*)d_in[3];
    const float* kv_w     = (const float*)d_in[4];
    const float* an_g     = (const float*)d_in[5];
    const float* an_b     = (const float*)d_in[6];
    const float* proj_w   = (const float*)d_in[7];
    const float* proj_b   = (const float*)d_in[8];
    const float* dconv_w  = (const float*)d_in[9];
    const float* dconv_b  = (const float*)d_in[10];
    const float* norm2_g  = (const float*)d_in[11];
    const float* norm2_b  = (const float*)d_in[12];
    const float* fc1_w    = (const float*)d_in[13];
    const float* fc1_b    = (const float*)d_in[14];
    const float* conv_w   = (const float*)d_in[15];
    const float* conv_b   = (const float*)d_in[16];
    const float* fc2_w    = (const float*)d_in[17];
    const float* fc2_b    = (const float*)d_in[18];
    float* outf = (float*)d_out;

    char* base = (char*)d_ws;
    size_t off = 0;
    auto take = [&](size_t bytes) { char* p = base + off; off += (bytes + 255) & ~(size_t)255; return p; };

    const size_t BN = (size_t)Bsz * Nn;        // 36864
    u16*   xn     = (u16*)take(BN * Cc * 2);
    u16*   qb     = (u16*)take(BN * Cc * 2);
    u16*   attnb  = (u16*)take(BN * Cc * 2);
    float* x1     = (float*)take(BN * Cc * 4);
    u16*   tb     = (u16*)take(BN * Cc * 2);
    u16*   t1b    = (u16*)take(BN * HID * 2);
    u16*   t2b    = (u16*)take(BN * HID * 2);
    float* poolr  = (float*)take((size_t)Bsz * Lpool * Cc * 4);
    float* pool2  = (float*)take((size_t)Bsz * Lpool * Cc * 4);
    u16*   pooln  = (u16*)take((size_t)Bsz * Lpool * Cc * 2);
    u16*   kvb    = (u16*)take((size_t)Bsz * Lpool * 1024 * 2);
    u16*   qwb    = (u16*)take((size_t)Cc * Cc * 2);
    u16*   kvwb   = (u16*)take((size_t)2 * Cc * Cc * 2);
    u16*   projwb = (u16*)take((size_t)Cc * Cc * 2);
    u16*   fc1wb  = (u16*)take((size_t)HID * Cc * 2);
    u16*   fc2wb  = (u16*)take((size_t)Cc * HID * 2);

    // 1) weights -> bf16
    cvt_bf16_kernel<<<(Cc * Cc + 255) / 256, 256, 0, stream>>>(q_w, qwb, Cc * Cc);
    cvt_bf16_kernel<<<(2 * Cc * Cc + 255) / 256, 256, 0, stream>>>(kv_w, kvwb, 2 * Cc * Cc);
    cvt_bf16_kernel<<<(Cc * Cc + 255) / 256, 256, 0, stream>>>(proj_w, projwb, Cc * Cc);
    cvt_bf16_kernel<<<(HID * Cc + 255) / 256, 256, 0, stream>>>(fc1_w, fc1wb, HID * Cc);
    cvt_bf16_kernel<<<(Cc * HID + 255) / 256, 256, 0, stream>>>(fc2_w, fc2wb, Cc * HID);

    // 2) LN1 -> xn (bf16)
    ln_kernel<<<(int)BN, 256, 0, stream>>>(x, norm1_g, norm1_b, xn);

    // 3) Q = xn @ q_w^T
    gemm_bf16_kernel<0><<<dim3(Cc / 128, (int)(BN / 128)), 256, 0, stream>>>(
        xn, qwb, nullptr, nullptr, qb, (int)BN, Cc, Cc);

    // 4) pooling path
    pool_sum_kernel<<<dim3(Lpool, Bsz), 256, 0, stream>>>(xn, poolr);
    pool_dw_kernel<<<dim3(Lpool, Bsz), 256, 0, stream>>>(poolr, dconv_w, dconv_b, pool2);
    ln_kernel<<<Bsz * Lpool, 256, 0, stream>>>(pool2, an_g, an_b, pooln);
    gemm_bf16_kernel<0><<<dim3(1024 / 128, (Bsz * Lpool + 127) / 128), 256, 0, stream>>>(
        pooln, kvwb, nullptr, nullptr, kvb, Bsz * Lpool, 1024, Cc);

    // 5) attention
    attn_kernel<<<dim3(Nn / 64, HEADS, Bsz), 128, 0, stream>>>(qb, kvb, attnb);

    // 6) proj + bias + residual(x) -> x1 (f32)
    gemm_bf16_kernel<2><<<dim3(Cc / 128, (int)(BN / 128)), 256, 0, stream>>>(
        attnb, projwb, proj_b, x, x1, (int)BN, Cc, Cc);

    // 7) LN2 -> tb, fc1 + bias + hardswish -> t1b (bf16)
    ln_kernel<<<(int)BN, 256, 0, stream>>>(x1, norm2_g, norm2_b, tb);
    gemm_bf16_kernel<1><<<dim3(HID / 128, (int)(BN / 128)), 256, 0, stream>>>(
        tb, fc1wb, fc1_b, nullptr, t1b, (int)BN, HID, Cc);

    // 8) depthwise 3x3 + bias + hardswish -> t2b
    dwconv_kernel<<<dim3(HID / 64, Hh / 4, Bsz), 256, 0, stream>>>(t1b, conv_w, conv_b, t2b);

    // 9) fc2 + bias + residual(x1) -> d_out (f32)
    gemm_bf16_kernel<2><<<dim3(Cc / 128, (int)(BN / 128)), 256, 0, stream>>>(
        t2b, fc2wb, fc2_b, x1, outf, (int)BN, Cc, HID);
}
